// HierKVRouter_22703197127136
// MI455X (gfx1250) — matrix operand order repaired and checked by reading, not compiled, not run on hardware
//
#include <hip/hip_runtime.h>
#include <hip/hip_bf16.h>

typedef _Float16 v4h  __attribute__((ext_vector_type(4)));
typedef _Float16 v8h  __attribute__((ext_vector_type(8)));
typedef _Float16 v16h __attribute__((ext_vector_type(16)));
typedef float    v4f  __attribute__((ext_vector_type(4)));
typedef float    v8f  __attribute__((ext_vector_type(8)));

#define EMBED   1024
#define NBUCKET 64
#define EPB     8
#define NEXP    512            // NBUCKET * EPB
#define NTILES  32             // NEXP / 16
#define AST     1032           // padded A row stride in halfs (1024 + 8 -> conflict-free ds_load_b128)
#define SST     528            // padded score row stride in floats (512 + 16)

// ---------------------------------------------------------------------------
// Kernel 1: L2-normalize the 512 expert keys, store f16 row-major in scratch.
// One wave per key row; vectorized float4 loads, v4h (b64) stores.
// ---------------------------------------------------------------------------
__global__ __launch_bounds__(256) void hkv_norm_keys(const float* __restrict__ ek,
                                                     _Float16* __restrict__ kh) {
  const int lane = threadIdx.x & 31;
  const int wv   = threadIdx.x >> 5;
  const int row  = blockIdx.x * 8 + wv;               // 0..511
  const size_t base = (size_t)row * EMBED;

  v4f r[8];
  float s = 0.f;
  #pragma unroll
  for (int j = 0; j < 8; ++j) {
    v4f v = *(const v4f*)(ek + base + j * 128 + lane * 4);   // global_load_b128
    r[j] = v;
    s += v.x * v.x + v.y * v.y + v.z * v.z + v.w * v.w;
  }
  #pragma unroll
  for (int off = 16; off > 0; off >>= 1) s += __shfl_xor(s, off, 32);
  const float scale = 1.0f / fmaxf(sqrtf(s), 1e-12f);
  #pragma unroll
  for (int j = 0; j < 8; ++j) {
    v4h o;
    o.x = (_Float16)(r[j].x * scale);
    o.y = (_Float16)(r[j].y * scale);
    o.z = (_Float16)(r[j].z * scale);
    o.w = (_Float16)(r[j].w * scale);
    *(v4h*)(kh + base + j * 128 + lane * 4) = o;             // global_store_b64
  }
}

// ---------------------------------------------------------------------------
// Kernel 2: one workgroup (8 wave32) per 16 tokens.
//  Phase A: normalize 16 h-rows into LDS f16 A-tile, gather buckets + tile mask
//  Phase B: waves compute needed 16x16 logit tiles via v_wmma_f32_16x16x32_f16
//  Phase C: per-token softmax(8) + top-2 + weight renorm, write outputs
// ---------------------------------------------------------------------------
__global__ __launch_bounds__(256) void hkv_route(const float* __restrict__ h,
                                                 const int*   __restrict__ op_id,
                                                 const _Float16* __restrict__ kh,
                                                 float* __restrict__ out,
                                                 int ntok) {
  __shared__ __align__(16) _Float16 Alds[16 * AST];   // 33,024 B
  __shared__ float    Sc[16 * SST];                   // 33,792 B
  __shared__ int      sbucket[16];
  __shared__ unsigned smask;

  const int tid     = threadIdx.x;
  const int lane    = tid & 31;
  const int wv      = tid >> 5;
  const int l15     = lane & 15;
  const int hi      = lane >> 4;
  const int tokbase = blockIdx.x * 16;

  if (tid == 0) smask = 0u;

  // ---- Phase A: normalize h rows (2 per wave) into LDS as f16 ----
  #pragma unroll
  for (int rr = 0; rr < 2; ++rr) {
    const int m = wv * 2 + rr;
    const size_t base = (size_t)(tokbase + m) * EMBED;
    v4f r[8];
    float s = 0.f;
    #pragma unroll
    for (int j = 0; j < 8; ++j) {
      v4f v = *(const v4f*)(h + base + j * 128 + lane * 4);  // global_load_b128
      r[j] = v;
      s += v.x * v.x + v.y * v.y + v.z * v.z + v.w * v.w;
    }
    #pragma unroll
    for (int off = 16; off > 0; off >>= 1) s += __shfl_xor(s, off, 32);
    const float scale = 1.0f / fmaxf(sqrtf(s), 1e-12f);
    #pragma unroll
    for (int j = 0; j < 8; ++j) {
      v4h o;
      o.x = (_Float16)(r[j].x * scale);
      o.y = (_Float16)(r[j].y * scale);
      o.z = (_Float16)(r[j].z * scale);
      o.w = (_Float16)(r[j].w * scale);
      *(v4h*)(&Alds[m * AST + j * 128 + lane * 4]) = o;      // ds_store_b64
    }
  }
  __syncthreads();          // smask init + A tile visible

  if (tid < 16) {
    int b = op_id[tokbase + tid];
    b = b < 0 ? 0 : (b > NBUCKET - 1 ? NBUCKET - 1 : b);
    sbucket[tid] = b;
    atomicOr(&smask, 1u << (b >> 1));   // N-tile covering experts [b*8, b*8+8)
  }
  __syncthreads();

  // ---- Phase B: WMMA over the distinct needed tiles, round-robin per wave ----
  const unsigned mask = smask;
  int c = 0;
  for (int t = 0; t < NTILES; ++t) {
    if (!((mask >> t) & 1u)) continue;
    if ((c++ & 7) != wv) continue;

    v8f acc = {};
    for (int k0 = 0; k0 < EMBED; k0 += 32) {
      // A fragment (16x32 f16): lane<16 -> M=lane, K {0..15};
      //                         lane>=16 -> M=lane-16, K {16..31}
      const _Float16* ap = &Alds[l15 * AST + k0 + (hi << 4)];
      v8h alo = *(const v8h*)ap;            // ds_load_b128
      v8h ahi = *(const v8h*)(ap + 8);      // ds_load_b128
      v16h a = __builtin_shufflevector(alo, ahi,
                 0, 1, 2, 3, 4, 5, 6, 7, 8, 9, 10, 11, 12, 13, 14, 15);

      // B fragment (32x16 f16): lane -> column N = t*16 + (lane&15),
      // contiguous K run (lane<16 -> K0..15, lane>=16 -> K16..31): 32B read
      const _Float16* bp = kh + ((size_t)(t * 16 + l15) * EMBED + k0 + (hi << 4));
      v16h b = *(const v16h*)bp;

      acc = __builtin_amdgcn_wmma_f32_16x16x32_f16(
                false, a, false, b, (short)0, acc, false, false);
    }
    // C layout: lane holds column N = lane&15 (+16 tile base), rows M = r + 8*(lane>=16)
    #pragma unroll
    for (int rr = 0; rr < 8; ++rr)
      Sc[(rr + (hi << 3)) * SST + t * 16 + l15] = acc[rr];
  }
  __syncthreads();

  // ---- Phase C: softmax over 8 experts, top-2, renormalized weights ----
  if (tid < 16) {
    const int b = sbucket[tid];
    const float* sr = &Sc[tid * SST + b * EPB];

    float mx = sr[0];
    #pragma unroll
    for (int j = 1; j < EPB; ++j) mx = fmaxf(mx, sr[j]);
    float p[EPB];
    float sum = 0.f;
    #pragma unroll
    for (int j = 0; j < EPB; ++j) { p[j] = __expf(sr[j] - mx); sum += p[j]; }
    const float inv = 1.0f / sum;

    int i0 = 0; float v0 = p[0] * inv;
    #pragma unroll
    for (int j = 1; j < EPB; ++j) {
      float pj = p[j] * inv;
      if (pj > v0) { v0 = pj; i0 = j; }
    }
    int i1 = (i0 == 0) ? 1 : 0; float v1 = p[i1] * inv;
    #pragma unroll
    for (int j = 0; j < EPB; ++j) {
      if (j == i0) continue;
      float pj = p[j] * inv;
      if (pj > v1) { v1 = pj; i1 = j; }
    }
    const float wden = 1.0f / (v0 + v1 + 1e-9f);

    const int g = tokbase + tid;
    out[(size_t)g * 2 + 0] = (float)(b * EPB + i0);      // gid[0]
    out[(size_t)g * 2 + 1] = (float)(b * EPB + i1);      // gid[1]
    float* ow = out + (size_t)ntok * 2;                  // w section
    ow[(size_t)g * 2 + 0] = v0 * wden;
    ow[(size_t)g * 2 + 1] = v1 * wden;
  }
}

// ---------------------------------------------------------------------------
extern "C" void kernel_launch(void* const* d_in, const int* in_sizes, int n_in,
                              void* d_out, int out_size, void* d_ws, size_t ws_size,
                              hipStream_t stream) {
  const float* h_ptr  = (const float*)d_in[0];   // (8,2048,1024) f32
  const int*   op_ptr = (const int*)d_in[1];     // (8,2048) int
  const float* ek_ptr = (const float*)d_in[2];   // (64,8,1024) f32
  float*       out    = (float*)d_out;           // gid(ntok,2) then w(ntok,2), flat f32

  const int ntok = in_sizes[0] / EMBED;          // 16384
  _Float16* kh = (_Float16*)d_ws;                // 512*1024 f16 = 1 MB scratch

  hipLaunchKernelGGL(hkv_norm_keys, dim3(NEXP / 8), dim3(256), 0, stream,
                     ek_ptr, kh);
  hipLaunchKernelGGL(hkv_route, dim3(ntok / 16), dim3(256), 0, stream,
                     h_ptr, op_ptr, kh, out, ntok);
}
